// MambaBlock_84602265796864
// MI455X (gfx1250) — matrix operand order repaired and checked by reading, not compiled, not run on hardware
//
#include <hip/hip_runtime.h>
#include <hip/hip_bf16.h>

// ---------------- problem constants ----------------
#define B_    2
#define L_    2048
#define DM_   1024          // d_model
#define DI_   2048          // d_inner
#define M_    4096          // B_*L_ rows
#define NPROJ_ 96           // dt_rank + 2*d_state
#define DTR_  64            // dt_rank
#define DS_   16            // d_state
#define NCH_  64            // scan chunks
#define CLEN_ 32            // steps per chunk (NCH_*CLEN_ == L_)

typedef __attribute__((ext_vector_type(16))) __bf16 v16bf;
typedef __attribute__((ext_vector_type(8)))  float  v8f;
typedef __attribute__((ext_vector_type(4)))  unsigned int v4u;
typedef __attribute__((ext_vector_type(4)))  int v4i;

union FragBF { struct { v4u lo, hi; } u; v16bf v; };

#define AS_GLOBAL __attribute__((address_space(1)))
#define AS_LOCAL  __attribute__((address_space(3)))

// gfx1250 async global->LDS path (ASYNCcnt-tracked), guarded so we always compile
#if defined(__has_builtin)
#if __has_builtin(__builtin_amdgcn_global_load_async_to_lds_b128)
#define USE_ASYNC_LDS 1
#endif
#endif

#if defined(USE_ASYNC_LDS)
// generic->global (AS1) is a bitwise identity on amdgcn
__device__ __forceinline__ AS_GLOBAL v4i* as_global_v4i(const void* p) {
  return (AS_GLOBAL v4i*)(unsigned long long)p;
}
// generic LDS pointer: low 32 bits are the LDS offset
__device__ __forceinline__ AS_LOCAL v4i* as_local_v4i(void* p) {
  return (AS_LOCAL v4i*)(unsigned int)(unsigned long long)p;
}
__device__ __forceinline__ void wait_async0() {
#if __has_builtin(__builtin_amdgcn_s_wait_asynccnt)
  __builtin_amdgcn_s_wait_asynccnt(0);
#else
  asm volatile("s_wait_asynccnt 0x0" ::: "memory");
#endif
}
#endif

__device__ __forceinline__ unsigned short f2bf(float f) {
  unsigned u = __builtin_bit_cast(unsigned, f);
  unsigned r = u + 0x7FFFu + ((u >> 16) & 1u);   // RNE
  return (unsigned short)(r >> 16);
}
__device__ __forceinline__ float bf2f(unsigned short h) {
  unsigned u = ((unsigned)h) << 16;
  return __builtin_bit_cast(float, u);
}

// ---------------- fp32 -> bf16 convert ----------------
__global__ void cvt_bf16_kernel(const float* __restrict__ src,
                                unsigned short* __restrict__ dst, int n) {
  int i = blockIdx.x * blockDim.x + threadIdx.x;
  if (i < n) dst[i] = f2bf(src[i]);
}

// ---------------- bf16 WMMA GEMM: C(f32, MxN) = A(MxK) * B(KxN) ----------------
// Assumes M % 128 == 0 and K % 32 == 0 (true for every call here).
// NGUARD=true additionally bounds-checks the N dimension (used for N=96).
#define BM 128
#define BN 128
#define KT 32
#define LDS_STRIDE 40   // 32 + 8 pad (uint16 elems) -> conflict-free b128 frag reads

template <bool NGUARD>
__global__ __launch_bounds__(256) void gemm_bf16_kernel(
    const unsigned short* __restrict__ A, const unsigned short* __restrict__ Bm,
    float* __restrict__ C, int M, int N, int K) {
  __shared__ unsigned short As[2][BM * LDS_STRIDE];
  __shared__ unsigned short Bs[2][BN * LDS_STRIDE];

  const int tid  = threadIdx.x;
  const int lane = tid & 31;
  const int wid  = tid >> 5;           // 8 waves
  const int waveM = wid >> 1;          // 0..3 -> 32-row stripes
  const int waveN = wid & 1;           // 0..1 -> 64-col stripes
  const int blockRow = blockIdx.y * BM;
  const int blockCol = blockIdx.x * BN;

  v8f acc[2][4];
  const v8f vzero = {0.f,0.f,0.f,0.f,0.f,0.f,0.f,0.f};
#pragma unroll
  for (int i = 0; i < 2; ++i)
#pragma unroll
    for (int j = 0; j < 4; ++j) acc[i][j] = vzero;

  // global->LDS load assignments
  const int aRow = tid >> 1;             // 0..127
  const int aColHalf = (tid & 1) * 16;   // 0 or 16
  const int bK2 = tid >> 4;              // 0..15 -> K pair {2*bK2, 2*bK2+1}
  const int bN8 = (tid & 15) * 8;        // 0..120

  // fragment lane decomposition (ISA 16-bit A / B layouts)
  const int mLane = lane & 15;
  const int kbA = (lane >> 4) * 8;       // A frag K sub-offset
  const int khB = (lane >> 4) * 8;       // B frag K sub-offset

  auto loadTiles = [&](int k0, int buf) {
    // ---- A tile: 128 x 32 row-major; no guards (M,K tile-aligned) ----
    {
      const unsigned short* src = A + (size_t)(blockRow + aRow) * K + k0 + aColHalf;
      unsigned short* dst = &As[buf][aRow * LDS_STRIDE + aColHalf];
#if defined(USE_ASYNC_LDS)
      __builtin_amdgcn_global_load_async_to_lds_b128(
          as_global_v4i(src), as_local_v4i(dst), 0, 0);
      __builtin_amdgcn_global_load_async_to_lds_b128(
          as_global_v4i(src + 8), as_local_v4i(dst + 8), 0, 0);
#else
      const v4u* p = (const v4u*)src;
      *(v4u*)dst = p[0];
      *(v4u*)(dst + 8) = p[1];
#endif
    }
    // ---- B tile: 32 x 128 global, stored transposed [n][k] in LDS ----
    // Each thread handles a K pair x 8 N, packing 2 K-values per b32 store.
    {
      int gK = k0 + 2 * bK2;
      int gN = blockCol + bN8;
      if (!NGUARD || gN + 8 <= N) {
        union { v4u v; unsigned short s[8]; } r0, r1;
        r0.v = *(const v4u*)(Bm + (size_t)gK * N + gN);
        r1.v = *(const v4u*)(Bm + (size_t)(gK + 1) * N + gN);
#pragma unroll
        for (int e = 0; e < 8; ++e) {
          unsigned pack = (unsigned)r0.s[e] | ((unsigned)r1.s[e] << 16);
          *(unsigned*)&Bs[buf][(bN8 + e) * LDS_STRIDE + 2 * bK2] = pack;
        }
      } else {
#pragma unroll
        for (int e = 0; e < 8; ++e) {
          unsigned short a0 = (gN + e < N) ? Bm[(size_t)gK * N + gN + e] : (unsigned short)0;
          unsigned short a1 = (gN + e < N) ? Bm[(size_t)(gK + 1) * N + gN + e] : (unsigned short)0;
          *(unsigned*)&Bs[buf][(bN8 + e) * LDS_STRIDE + 2 * bK2] =
              (unsigned)a0 | ((unsigned)a1 << 16);
        }
      }
    }
  };

  const int nk = K / KT;
  loadTiles(0, 0);
#if defined(USE_ASYNC_LDS)
  wait_async0();
#endif
  __syncthreads();

  for (int kt = 0; kt < nk; ++kt) {
    const int cur = kt & 1;
    if (kt + 1 < nk) loadTiles((kt + 1) * KT, cur ^ 1);
    if (kt + 2 < nk)
      __builtin_prefetch(A + (size_t)(blockRow + aRow) * K + (kt + 2) * KT + aColHalf, 0, 1);

    // ---- fragments from buffer `cur` ----
    v16bf aF[2], bF[4];
#pragma unroll
    for (int mf = 0; mf < 2; ++mf) {
      int m = waveM * 32 + mf * 16 + mLane;
      const unsigned short* p = &As[cur][m * LDS_STRIDE];
      FragBF f;
      f.u.lo = *(const v4u*)(p + kbA);        // K = kb..kb+7
      f.u.hi = *(const v4u*)(p + 16 + kbA);   // K = 16+kb..16+kb+7
      aF[mf] = f.v;
    }
#pragma unroll
    for (int nf = 0; nf < 4; ++nf) {
      int n = waveN * 64 + nf * 16 + mLane;
      const unsigned short* p = &Bs[cur][n * LDS_STRIDE + khB];
      FragBF f;
      f.u.lo = *(const v4u*)(p);              // K = kb..kb+7
      f.u.hi = *(const v4u*)(p + 16);         // K = 16+kb..16+kb+7
      bF[nf] = f.v;
    }
#pragma unroll
    for (int mf = 0; mf < 2; ++mf)
#pragma unroll
      for (int nf = 0; nf < 4; ++nf)
        acc[mf][nf] = __builtin_amdgcn_wmma_f32_16x16x32_bf16(
            false, aF[mf], false, bF[nf], (short)0, acc[mf][nf], false, false);

#if defined(USE_ASYNC_LDS)
    wait_async0();          // next tile's async A loads complete before barrier
#endif
    __syncthreads();
  }

  // ---- epilogue: C/D layout: VGPR r -> M = r (lanes 0-15) / 8+r (lanes 16-31)
  const int rBase = (lane >> 4) * 8;
  const int cLane = lane & 15;
#pragma unroll
  for (int mf = 0; mf < 2; ++mf)
#pragma unroll
    for (int nf = 0; nf < 4; ++nf) {
      int col = blockCol + waveN * 64 + nf * 16 + cLane;
      if (NGUARD && col >= N) continue;
#pragma unroll
      for (int r = 0; r < 8; ++r) {
        int row = blockRow + waveM * 32 + mf * 16 + rBase + r;
        C[(size_t)row * N + col] = acc[mf][nf][r];
      }
    }
}

// ---------------- causal depthwise conv(4) + bias + SiLU -> bf16 ----------------
__global__ void conv_silu_kernel(const float* __restrict__ xz,
                                 const float* __restrict__ conv_w,
                                 const float* __restrict__ conv_b,
                                 unsigned short* __restrict__ xcb) {
  int idx = blockIdx.x * blockDim.x + threadIdx.x;   // over B_*L_*DI_
  if (idx >= B_ * L_ * DI_) return;
  int d = idx % DI_;
  int t = (idx / DI_) % L_;
  int b = idx / (DI_ * L_);
  float acc = conv_b[d];
#pragma unroll
  for (int j = 0; j < 4; ++j) {
    int tt = t - 3 + j;
    if (tt >= 0)
      acc += conv_w[j * DI_ + d] * xz[((size_t)(b * L_ + tt)) * (2 * DI_) + d];
  }
  float s = acc / (1.f + __expf(-acc));
  xcb[idx] = f2bf(s);
}

// ---------------- extract dt_low (first 64 cols of proj) -> bf16 ----------------
__global__ void extract_dtlow_kernel(const float* __restrict__ proj,
                                     unsigned short* __restrict__ dtlb) {
  int idx = blockIdx.x * blockDim.x + threadIdx.x;   // over M_*DTR_
  if (idx >= M_ * DTR_) return;
  int r = idx >> 6, c = idx & 63;
  dtlb[idx] = f2bf(proj[(size_t)r * NPROJ_ + c]);
}

// ---------------- dt = softplus(dt_pre + b_dt), in place ----------------
__global__ void dt_softplus_kernel(float* __restrict__ dt,
                                   const float* __restrict__ b_dt) {
  int idx = blockIdx.x * blockDim.x + threadIdx.x;   // over M_*DI_
  if (idx >= M_ * DI_) return;
  float v = dt[idx] + b_dt[idx % DI_];
  dt[idx] = (v > 20.f) ? v : log1pf(__expf(v));
}

// ---------------- chunked selective scan, pass 1: per-chunk (prod a, q) ----------------
__global__ __launch_bounds__(256) void scan_pass1_kernel(
    const float* __restrict__ dt, const unsigned short* __restrict__ xcb,
    const float* __restrict__ proj, const float* __restrict__ A_log,
    float* __restrict__ hP, float* __restrict__ hQ) {
  int gid = blockIdx.x * 256 + threadIdx.x;           // (b*NCH_+c)*DI_+d
  if (gid >= B_ * NCH_ * DI_) return;
  int d = gid % DI_;
  int c = (gid / DI_) % NCH_;
  int b = gid / (DI_ * NCH_);
  float A[DS_], P[DS_], Q[DS_];
#pragma unroll
  for (int s = 0; s < DS_; ++s) { A[s] = -__expf(A_log[d * DS_ + s]); P[s] = 1.f; Q[s] = 0.f; }
  for (int t = c * CLEN_; t < (c + 1) * CLEN_; ++t) {
    size_t row = (size_t)(b * L_ + t);
    float dtv = dt[row * DI_ + d];
    float xv  = bf2f(xcb[row * DI_ + d]);
    float dx  = dtv * xv;
    const float* pr = proj + row * NPROJ_;
#pragma unroll
    for (int s = 0; s < DS_; ++s) {
      float a = __expf(dtv * A[s]);
      Q[s] = a * Q[s] + dx * pr[DTR_ + s];
      P[s] *= a;
    }
  }
#pragma unroll
  for (int s = 0; s < DS_; ++s) { hP[(size_t)gid * DS_ + s] = P[s]; hQ[(size_t)gid * DS_ + s] = Q[s]; }
}

// ---------------- pass 2: sequential chunk combine (length NCH_) ----------------
__global__ void scan_pass2_kernel(const float* __restrict__ hP,
                                  const float* __restrict__ hQ,
                                  float* __restrict__ hInit) {
  int gid = blockIdx.x * 256 + threadIdx.x;           // over B_*DI_*DS_
  if (gid >= B_ * DI_ * DS_) return;
  int s = gid & 15;
  int d = (gid >> 4) % DI_;
  int b = gid / (DS_ * DI_);
  float h = 0.f;
  for (int c = 0; c < NCH_; ++c) {
    size_t i = (((size_t)(b * NCH_ + c) * DI_) + d) * DS_ + s;
    hInit[i] = h;
    h = hP[i] * h + hQ[i];
  }
}

// ---------------- pass 3: replay chunk, emit gated y (bf16) ----------------
__global__ __launch_bounds__(256) void scan_pass3_kernel(
    const float* __restrict__ dt, const unsigned short* __restrict__ xcb,
    const float* __restrict__ proj, const float* __restrict__ xz,
    const float* __restrict__ A_log, const float* __restrict__ Dp,
    const float* __restrict__ hInit, unsigned short* __restrict__ yb) {
  int gid = blockIdx.x * 256 + threadIdx.x;           // (b*NCH_+c)*DI_+d
  if (gid >= B_ * NCH_ * DI_) return;
  int d = gid % DI_;
  int c = (gid / DI_) % NCH_;
  int b = gid / (DI_ * NCH_);
  float A[DS_], h[DS_];
#pragma unroll
  for (int s = 0; s < DS_; ++s) {
    A[s] = -__expf(A_log[d * DS_ + s]);
    h[s] = hInit[(size_t)gid * DS_ + s];
  }
  const float Dv = Dp[d];
  for (int t = c * CLEN_; t < (c + 1) * CLEN_; ++t) {
    size_t row = (size_t)(b * L_ + t);
    float dtv = dt[row * DI_ + d];
    float xv  = bf2f(xcb[row * DI_ + d]);
    float zv  = xz[row * (2 * DI_) + DI_ + d];
    float dx  = dtv * xv;
    const float* pr = proj + row * NPROJ_;
    float y = 0.f;
#pragma unroll
    for (int s = 0; s < DS_; ++s) {
      h[s] = __expf(dtv * A[s]) * h[s] + dx * pr[DTR_ + s];
      y += h[s] * pr[DTR_ + DS_ + s];
    }
    y += xv * Dv;
    y *= zv / (1.f + __expf(-zv));        // * silu(z)
    yb[row * DI_ + d] = f2bf(y);
  }
}

// ---------------- host orchestration ----------------
extern "C" void kernel_launch(void* const* d_in, const int* in_sizes, int n_in,
                              void* d_out, int out_size, void* d_ws, size_t ws_size,
                              hipStream_t stream) {
  (void)in_sizes; (void)n_in; (void)out_size; (void)ws_size;
  const float* hs     = (const float*)d_in[0];
  const float* W_in   = (const float*)d_in[1];
  const float* conv_w = (const float*)d_in[2];
  const float* conv_b = (const float*)d_in[3];
  const float* W_x    = (const float*)d_in[4];
  const float* W_dt   = (const float*)d_in[5];
  const float* b_dt   = (const float*)d_in[6];
  const float* A_log  = (const float*)d_in[7];
  const float* Dp     = (const float*)d_in[8];
  const float* W_out  = (const float*)d_in[9];
  float* out = (float*)d_out;

  char* ws = (char*)d_ws;
  size_t off = 0;
  auto alloc = [&](size_t bytes) -> char* {
    char* p = ws + off;
    off += (bytes + 255) & ~(size_t)255;
    return p;
  };
  unsigned short* hsb   = (unsigned short*)alloc((size_t)M_ * DM_ * 2);       // 8 MB
  unsigned short* Winb  = (unsigned short*)alloc((size_t)DM_ * 2 * DI_ * 2);  // 8 MB
  float*          xz    = (float*)alloc((size_t)M_ * 2 * DI_ * 4);            // 64 MB
  unsigned short* xcb   = (unsigned short*)alloc((size_t)M_ * DI_ * 2);       // 16 MB
  unsigned short* Wxb   = (unsigned short*)alloc((size_t)DI_ * NPROJ_ * 2);
  float*          proj  = (float*)alloc((size_t)M_ * NPROJ_ * 4);
  unsigned short* dtlb  = (unsigned short*)alloc((size_t)M_ * DTR_ * 2);
  unsigned short* Wdtb  = (unsigned short*)alloc((size_t)DTR_ * DI_ * 2);
  float*          dtf   = (float*)alloc((size_t)M_ * DI_ * 4);                // 32 MB
  unsigned short* yb    = (unsigned short*)alloc((size_t)M_ * DI_ * 2);       // 16 MB
  unsigned short* Woutb = (unsigned short*)alloc((size_t)DI_ * DM_ * 2);      // 4 MB
  float*          hQ    = (float*)alloc((size_t)B_ * NCH_ * DI_ * DS_ * 4);   // 16.8 MB
  float*          hInit = (float*)alloc((size_t)B_ * NCH_ * DI_ * DS_ * 4);   // 16.8 MB
  // hP reuses the hsb+Winb region (dead after GEMM 1); sizes match exactly (16 MB)
  float* hP = (float*)(ws + 0);

  const int CT = 256;
  auto cdiv = [](int a, int b) { return (a + b - 1) / b; };

  // 1) bf16 conversions for GEMM operands
  cvt_bf16_kernel<<<cdiv(M_ * DM_, CT), CT, 0, stream>>>(hs, hsb, M_ * DM_);
  cvt_bf16_kernel<<<cdiv(DM_ * 2 * DI_, CT), CT, 0, stream>>>(W_in, Winb, DM_ * 2 * DI_);

  // 2) xz = hs @ W_in  (4096 x 4096, K=1024) -- full tiles
  gemm_bf16_kernel<false><<<dim3(cdiv(2 * DI_, BN), cdiv(M_, BM)), CT, 0, stream>>>(
      hsb, Winb, xz, M_, 2 * DI_, DM_);

  // 3) depthwise conv + SiLU -> x (bf16)
  conv_silu_kernel<<<cdiv(B_ * L_ * DI_, CT), CT, 0, stream>>>(xz, conv_w, conv_b, xcb);

  // 4) proj = x @ W_x  (4096 x 96, K=2048) -- N guarded
  cvt_bf16_kernel<<<cdiv(DI_ * NPROJ_, CT), CT, 0, stream>>>(W_x, Wxb, DI_ * NPROJ_);
  gemm_bf16_kernel<true><<<dim3(cdiv(NPROJ_, BN), cdiv(M_, BM)), CT, 0, stream>>>(
      xcb, Wxb, proj, M_, NPROJ_, DI_);

  // 5) dt_pre = dt_low @ W_dt  (4096 x 2048, K=64), then softplus(+b_dt)
  extract_dtlow_kernel<<<cdiv(M_ * DTR_, CT), CT, 0, stream>>>(proj, dtlb);
  cvt_bf16_kernel<<<cdiv(DTR_ * DI_, CT), CT, 0, stream>>>(W_dt, Wdtb, DTR_ * DI_);
  gemm_bf16_kernel<false><<<dim3(cdiv(DI_, BN), cdiv(M_, BM)), CT, 0, stream>>>(
      dtlb, Wdtb, dtf, M_, DI_, DTR_);
  dt_softplus_kernel<<<cdiv(M_ * DI_, CT), CT, 0, stream>>>(dtf, b_dt);

  // 6) chunked selective scan (3 passes) + gating, emits yb (bf16)
  scan_pass1_kernel<<<cdiv(B_ * NCH_ * DI_, CT), CT, 0, stream>>>(
      dtf, xcb, proj, A_log, hP, hQ);
  scan_pass2_kernel<<<cdiv(B_ * DI_ * DS_, CT), CT, 0, stream>>>(hP, hQ, hInit);
  scan_pass3_kernel<<<cdiv(B_ * NCH_ * DI_, CT), CT, 0, stream>>>(
      dtf, xcb, proj, xz, A_log, Dp, hInit, yb);

  // 7) out = y @ W_out  (4096 x 1024, K=2048) -- full tiles
  cvt_bf16_kernel<<<cdiv(DM_ * DI_, CT), CT, 0, stream>>>(W_out, Woutb, DI_ * DM_);
  gemm_bf16_kernel<false><<<dim3(cdiv(DM_, BN), cdiv(M_, BM)), CT, 0, stream>>>(
      yb, Woutb, out, M_, DM_, DI_);
}
